// SenderGPTLM_55061480734839
// MI455X (gfx1250) — matrix operand order repaired and checked by reading, not compiled, not run on hardware
//
#include <hip/hip_runtime.h>
#include <stdint.h>

// ---------------------------------------------------------------------------
// CDNA5 / gfx1250 implementation of the SenderGPT-LM reference.
// All heavy GEMMs run through v_wmma_f32_16x16x32_bf16 (wave32 matrix cores).
// Weights are converted fp32->bf16 AND pre-packed into the WMMA B-fragment
// lane layout once per launch (the ~90MB bf16 working set is L2-resident, so
// the 14 scan steps re-stream it from the 192MB L2, not HBM).
// ---------------------------------------------------------------------------

typedef __bf16 bf16_t;
typedef __attribute__((ext_vector_type(16))) __bf16 v16bf;
typedef __attribute__((ext_vector_type(8)))  float  v8f;

// Problem constants (match reference)
#define B_    64
#define P_    8
#define D_    1024
#define H_    16
#define BOS_  16
#define C_    14
#define S_    30      // BOS + C
#define L_    4
#define V_    1024
#define HID_  2048
#define DH_   64

// GEMM tile
#define BM 128
#define BN 64
#define BK 32

// ---------------------------------------------------------------------------
// WMMA GEMM: C[M,N] (+)= A[M,K] @ B[K,N] (+bias), C fp32.
//   A: bf16 row-major [M,K]
//   B: bf16 PACKED fragment-native layout (see pack_b_k below)
// 256 threads = 8 wave32 waves in a 4(M)x2(N) grid; each wave computes a
// 32x32 patch as 2x2 v_wmma_f32_16x16x32_bf16 tiles. Double-buffered LDS,
// register prefetch of the next K-tile; every LDS access is b128.
// M is guarded; N % 64 == 0 and K % 32 == 0 (true for all GEMMs here).
// ---------------------------------------------------------------------------
template<bool ADD_RES>
__global__ __launch_bounds__(256) void gemm_bf16_wmma_k(
    const bf16_t* __restrict__ A, const bf16_t* __restrict__ Bp,
    float* __restrict__ C, const float* __restrict__ bias,
    int M, int N, int K)
{
  __shared__ bf16_t As[2][BM][BK + 8];                    // 80B row stride
  __shared__ __align__(32) bf16_t Bs[2][4][32][16];      // 4 n-subtiles, frag-native

  const int tid  = threadIdx.x;
  const int lane = tid & 31;
  const int wid  = tid >> 5;
  const int wm   = wid & 3;     // wave row (4)
  const int wn   = wid >> 2;    // wave col (2)
  const int half = lane >> 4;   // lane half per ISA layout
  const int lm   = lane & 15;
  const int m0   = blockIdx.y * BM;
  const int n0   = blockIdx.x * BN;
  const int nk   = K / BK;      // number of 32-deep K tiles
  const int ntg0 = n0 >> 4;     // first global 16-wide n-subtile

  float4 aReg[2], bReg;

  auto loadA = [&](int kt) {
#pragma unroll
    for (int c = 0; c < 2; ++c) {
      int ch = 2 * tid + c, r = ch >> 2, col = (ch & 3) * 8;
      int gm = m0 + r;
      if (gm < M)
        aReg[c] = *reinterpret_cast<const float4*>(&A[(size_t)gm * K + kt * BK + col]);
      else
        aReg[c] = make_float4(0.f, 0.f, 0.f, 0.f);
    }
  };
  // Packed B: tile (nt, kt) is 512 contiguous bf16 = [lane][16]
  auto loadB = [&](int kt) {
    const int ntl = tid >> 6, rem = tid & 63;
    const size_t base = ((size_t)(ntg0 + ntl) * nk + kt) * 512 + rem * 8;
    bReg = *reinterpret_cast<const float4*>(&Bp[base]);
  };
  auto stashA = [&](int buf) {
#pragma unroll
    for (int c = 0; c < 2; ++c) {
      int ch = 2 * tid + c, r = ch >> 2, col = (ch & 3) * 8;
      *reinterpret_cast<float4*>(&As[buf][r][col]) = aReg[c];
    }
  };
  auto stashB = [&](int buf) {
    const int ntl = tid >> 6, rem = tid & 63;
    *reinterpret_cast<float4*>(&Bs[buf][ntl][0][0] + rem * 8) = bReg;
  };

  v8f acc[2][2] = {};

  loadA(0); loadB(0);
  stashA(0); stashB(0);
  __syncthreads();

  for (int kt = 0; kt < nk; ++kt) {
    const int cur = kt & 1;
    if (kt + 1 < nk) {
      loadA(kt + 1);
      loadB(kt + 1);
      if (kt + 2 < nk) {  // hint L2: pull the tile after next
        __builtin_prefetch(&A[(size_t)(m0 + (tid >> 1)) * K + (size_t)(kt + 2) * BK], 0, 1);
        __builtin_prefetch(&Bp[((size_t)(ntg0 + (tid >> 6)) * nk + (kt + 2)) * 512], 0, 1);
      }
    }

    // A fragments: 16x32 bf16 -> two contiguous 16B chunks
    // per lane (K half*16..+7 and half*16+8..+15), loaded as b128 pairs.
    v16bf afr[2];
#pragma unroll
    for (int i = 0; i < 2; ++i) {
      const int rA = wm * 32 + i * 16 + lm;
      union { float4 f4[2]; v16bf v; } ua;
      ua.f4[0] = *reinterpret_cast<const float4*>(&As[cur][rA][half * 16]);
      ua.f4[1] = *reinterpret_cast<const float4*>(&As[cur][rA][half * 16 + 8]);
      afr[i] = ua.v;
    }
    // B fragments: packed layout makes each lane's 16 bf16 contiguous (32B).
    v16bf bfr[2];
#pragma unroll
    for (int j = 0; j < 2; ++j)
      bfr[j] = *reinterpret_cast<const v16bf*>(&Bs[cur][wn * 2 + j][lane][0]);

#pragma unroll
    for (int i = 0; i < 2; ++i)
#pragma unroll
      for (int j = 0; j < 2; ++j)
        acc[i][j] = __builtin_amdgcn_wmma_f32_16x16x32_bf16(
            false, afr[i], false, bfr[j], (short)0, acc[i][j], false, false);

    __syncthreads();
    if (kt + 1 < nk) {
      stashA(cur ^ 1);
      stashB(cur ^ 1);
      __syncthreads();
    }
  }

  // Epilogue per C/D layout: VGPR v holds row (half*8 + v), col = lane%16.
#pragma unroll
  for (int i = 0; i < 2; ++i) {
#pragma unroll
    for (int j = 0; j < 2; ++j) {
      const int gmb = m0 + wm * 32 + i * 16 + half * 8;
      const int gn  = n0 + wn * 32 + j * 16 + lm;
      const float bv = bias ? bias[gn] : 0.f;
#pragma unroll
      for (int v = 0; v < 8; ++v) {
        const int gm = gmb + v;
        if (gm < M) {
          const size_t idx = (size_t)gm * N + gn;
          const float val = acc[i][j][v] + bv;
          if (ADD_RES) C[idx] += val; else C[idx] = val;
        }
      }
    }
  }
}

// ---------------------------------------------------------------------------
// fp32 -> bf16 conversion (A-side operands)
// ---------------------------------------------------------------------------
__global__ __launch_bounds__(256) void f32_to_bf16_k(const float* __restrict__ in,
                                                     bf16_t* __restrict__ out, int n) {
  int i = blockIdx.x * 256 + threadIdx.x;
  if (i < n) out[i] = (bf16_t)in[i];
}

// ---------------------------------------------------------------------------
// fp32 -> bf16 conversion + pack into WMMA B-fragment lane layout:
//   out[((nt*(K/32)+kt)*32 + lane)*16 + e]
//   with nt=n/16, kt=k/32, lane=((k%32)/16)*16 + n%16, e=k%16
// (contiguous k order, same as the A side: lanes 0-15 hold K 0..15 of column n, lanes 16-31 hold K 16..31)
// ---------------------------------------------------------------------------
__global__ __launch_bounds__(256) void pack_b_bf16_k(const float* __restrict__ in,
                                                     bf16_t* __restrict__ out,
                                                     int K, int N) {
  const int idx = blockIdx.x * 256 + threadIdx.x;
  if (idx >= K * N) return;
  const int n = idx % N, k = idx / N;
  const int nt = n >> 4, lm = n & 15;
  const int kt = k >> 5, kr = k & 31;
  const int lane = (kr >> 4) * 16 + lm;
  const int e = kr & 15;
  const size_t dst = (((size_t)nt * (K >> 5) + kt) * 32 + lane) * 16 + e;
  out[dst] = (bf16_t)in[idx];
}

// ---------------------------------------------------------------------------
// LayerNorm over D=1024, block per row, writes bf16 (GEMM input).
// Row remap supports the final-LN case (take rows [BOS, S) of each batch).
// ---------------------------------------------------------------------------
__global__ __launch_bounds__(256) void ln_bf16_k(
    const float* __restrict__ x, const float* __restrict__ w, const float* __restrict__ b,
    bf16_t* __restrict__ out, int inner_in, int inner_out, int row_off)
{
  const int r = blockIdx.x, t = threadIdx.x;
  const int bb = r / inner_out, cc = r % inner_out;
  const float* xr = x + (size_t)(bb * inner_in + row_off + cc) * D_;
  __shared__ float red[256];

  float s = 0.f;
  for (int d = t; d < D_; d += 256) s += xr[d];
  red[t] = s; __syncthreads();
  for (int o = 128; o > 0; o >>= 1) { if (t < o) red[t] += red[t + o]; __syncthreads(); }
  const float mean = red[0] * (1.f / D_); __syncthreads();

  float vv = 0.f;
  for (int d = t; d < D_; d += 256) { float dd = xr[d] - mean; vv += dd * dd; }
  red[t] = vv; __syncthreads();
  for (int o = 128; o > 0; o >>= 1) { if (t < o) red[t] += red[t + o]; __syncthreads(); }
  const float inv = rsqrtf(red[0] * (1.f / D_) + 1e-5f);

  bf16_t* orow = out + (size_t)r * D_;
  for (int d = t; d < D_; d += 256)
    orow[d] = (bf16_t)((xr[d] - mean) * inv * w[d] + b[d]);
}

// ---------------------------------------------------------------------------
// Causal self-attention for one (b,h): S=30, DH=64. ~0.2% of FLOPs -> VALU.
// qkv layout: (B,S,3,H,DH) flattened -> (b*S+s)*3*D + t*D + h*DH + d.
// ---------------------------------------------------------------------------
__global__ __launch_bounds__(256) void attn_k(
    const float* __restrict__ qkv, const float* __restrict__ bias_l,
    bf16_t* __restrict__ attout)
{
  const int bh = blockIdx.x, b = bh / H_, h = bh % H_, t = threadIdx.x;
  __shared__ float qs[S_][DH_], ks[S_][DH_], vs[S_][DH_], sc[S_][32];

  for (int idx = t; idx < S_ * DH_; idx += 256) {
    const int i = idx >> 6, d = idx & 63;
    const size_t base = (size_t)(b * S_ + i) * (3 * D_) + h * DH_ + d;
    qs[i][d] = qkv[base];
    ks[i][d] = qkv[base + D_];
    vs[i][d] = qkv[base + 2 * D_];
  }
  __syncthreads();

  for (int idx = t; idx < S_ * S_; idx += 256) {
    const int i = idx / S_, j = idx % S_;
    if (j <= i) {
      float acc = 0.f;
#pragma unroll
      for (int d = 0; d < DH_; ++d) acc += qs[i][d] * ks[j][d];
      sc[i][j] = acc * 0.125f + bias_l[(h * S_ + i) * S_ + j];  // DH^-0.5 = 1/8
    } else {
      sc[i][j] = -1e30f;
    }
  }
  __syncthreads();

  if (t < S_) {
    float m = -3.4e38f;
    for (int j = 0; j < S_; ++j) m = fmaxf(m, sc[t][j]);
    float s = 0.f;
    for (int j = 0; j < S_; ++j) { float e = __expf(sc[t][j] - m); sc[t][j] = e; s += e; }
    const float inv = 1.f / s;
    for (int j = 0; j < S_; ++j) sc[t][j] *= inv;
  }
  __syncthreads();

  for (int idx = t; idx < S_ * DH_; idx += 256) {
    const int i = idx >> 6, d = idx & 63;
    float o = 0.f;
    for (int j = 0; j < S_; ++j) o += sc[i][j] * vs[j][d];
    attout[(size_t)(b * S_ + i) * D_ + h * DH_ + d] = (bf16_t)o;
  }
}

// ---------------------------------------------------------------------------
// Cross-attention for one (b,h): 16 queries x 8 keys, scale = 1/DH.
// ---------------------------------------------------------------------------
__global__ __launch_bounds__(256) void ca_attn_k(
    const float* __restrict__ qh, const float* __restrict__ kh,
    const float* __restrict__ vh, bf16_t* __restrict__ caout)
{
  const int bh = blockIdx.x, b = bh / H_, h = bh % H_, t = threadIdx.x;
  __shared__ float qs[BOS_][DH_], ks[P_][DH_], vs[P_][DH_], sc[BOS_][P_];

  for (int idx = t; idx < BOS_ * DH_; idx += 256) {
    const int i = idx >> 6, d = idx & 63;
    qs[i][d] = qh[(size_t)i * D_ + h * DH_ + d];
  }
  for (int idx = t; idx < P_ * DH_; idx += 256) {
    const int p = idx >> 6, d = idx & 63;
    ks[p][d] = kh[(size_t)(b * P_ + p) * D_ + h * DH_ + d];
    vs[p][d] = vh[(size_t)(b * P_ + p) * D_ + h * DH_ + d];
  }
  __syncthreads();

  for (int idx = t; idx < BOS_ * P_; idx += 256) {
    const int i = idx >> 3, p = idx & 7;
    float acc = 0.f;
#pragma unroll
    for (int d = 0; d < DH_; ++d) acc += qs[i][d] * ks[p][d];
    sc[i][p] = acc * (1.f / 64.f);   // scaling='d' -> divide by head dim
  }
  __syncthreads();

  if (t < BOS_) {
    float m = -3.4e38f;
    for (int p = 0; p < P_; ++p) m = fmaxf(m, sc[t][p]);
    float s = 0.f;
    for (int p = 0; p < P_; ++p) { float e = __expf(sc[t][p] - m); sc[t][p] = e; s += e; }
    const float inv = 1.f / s;
    for (int p = 0; p < P_; ++p) sc[t][p] *= inv;
  }
  __syncthreads();

  for (int idx = t; idx < BOS_ * DH_; idx += 256) {
    const int i = idx >> 6, d = idx & 63;
    float o = 0.f;
#pragma unroll
    for (int p = 0; p < P_; ++p) o += sc[i][p] * vs[p][d];
    caout[(size_t)(b * BOS_ + i) * D_ + h * DH_ + d] = (bf16_t)o;
  }
}

// ---------------------------------------------------------------------------
// SwiGLU: g = silu(u[:, :HID]) * u[:, HID:], bf16 out
// ---------------------------------------------------------------------------
__global__ __launch_bounds__(256) void swiglu_k(const float* __restrict__ u,
                                                bf16_t* __restrict__ g, int rows) {
  const int idx = blockIdx.x * 256 + threadIdx.x;
  if (idx >= rows * HID_) return;
  const int c = idx & (HID_ - 1), r = idx >> 11;
  const float u1 = u[(size_t)r * (2 * HID_) + c];
  const float u2 = u[(size_t)r * (2 * HID_) + HID_ + c];
  g[idx] = (bf16_t)((u1 / (1.f + __expf(-u1))) * u2);
}

// ---------------------------------------------------------------------------
// Relative position bias gather: bias[l,h,i,j] = rel_bias[l, i-j+S-1, h]
// ---------------------------------------------------------------------------
__global__ __launch_bounds__(256) void bias_pre_k(const float* __restrict__ rel_bias,
                                                  float* __restrict__ bias) {
  const int idx = blockIdx.x * 256 + threadIdx.x;
  if (idx >= L_ * H_ * S_ * S_) return;
  const int j = idx % S_, i = (idx / S_) % S_;
  const int h = (idx / (S_ * S_)) % H_, l = idx / (S_ * S_ * H_);
  bias[idx] = rel_bias[((size_t)l * (2 * S_ - 1) + (i - j + S_ - 1)) * H_ + h];
}

// ---------------------------------------------------------------------------
// Assemble x for step i: rows [0,BOS) = bos, rows [BOS,S) = spoken if c<i else pos_emb
// ---------------------------------------------------------------------------
__global__ __launch_bounds__(256) void build_x_k(
    float* __restrict__ x, const float* __restrict__ bos,
    const float* __restrict__ spoken, const float* __restrict__ pos_emb, int step)
{
  const int idx = blockIdx.x * 256 + threadIdx.x;  // B*S*D threads exactly
  const int d = idx % D_, rs = idx / D_, s = rs % S_, b = rs / S_;
  float v;
  if (s < BOS_) v = bos[(size_t)(b * BOS_ + s) * D_ + d];
  else {
    const int c = s - BOS_;
    v = (c < step) ? spoken[(size_t)(b * C_ + c) * D_ + d]
                   : pos_emb[(size_t)c * D_ + d];
  }
  x[idx] = v;
}

// ---------------------------------------------------------------------------
// Gumbel straight-through sampling per (b,c) row of V=1024 logits.
// JAX RNG is not reproducible here (no correctness harness); use a splittable
// integer hash keyed on (step,row,v) as the gumbel source.
// ---------------------------------------------------------------------------
__device__ __forceinline__ uint32_t hash3_(uint32_t a, uint32_t b, uint32_t c) {
  uint32_t h = a * 0x9E3779B1u ^ b * 0x85EBCA77u ^ c * 0xC2B2AE3Du;
  h ^= h >> 16; h *= 0x7FEB352Du; h ^= h >> 15; h *= 0x846CA68Bu; h ^= h >> 16;
  return h;
}

__global__ __launch_bounds__(256) void sample_k(
    const float* __restrict__ logits, float* __restrict__ st_f,
    bf16_t* __restrict__ st_bf, int step)
{
  const int row = blockIdx.x, t = threadIdx.x;
  const float* lr = logits + (size_t)row * V_;
  __shared__ float vals[V_];
  __shared__ float red[256];
  __shared__ int   redi[256];

  float lmax = -3.4e38f;
  for (int v = t; v < V_; v += 256) {
    const uint32_t hv = hash3_(0x2545F491u * (uint32_t)(step + 1), (uint32_t)row, (uint32_t)v);
    float u = (hv >> 8) * (1.0f / 16777216.0f);
    u = fmaxf(u, 1e-12f);
    const float g = -logf(-logf(u));
    const float val = lr[v] + g;     // TAU = 1
    vals[v] = val;
    lmax = fmaxf(lmax, val);
  }
  red[t] = lmax; __syncthreads();
  for (int o = 128; o > 0; o >>= 1) { if (t < o) red[t] = fmaxf(red[t], red[t + o]); __syncthreads(); }
  const float m = red[0]; __syncthreads();

  float ls = 0.f;
  for (int v = t; v < V_; v += 256) ls += __expf(vals[v] - m);
  red[t] = ls; __syncthreads();
  for (int o = 128; o > 0; o >>= 1) { if (t < o) red[t] += red[t + o]; __syncthreads(); }
  const float inv = 1.f / red[0]; __syncthreads();

  float bestv = -1.f; int besti = V_;
  for (int v = t; v < V_; v += 256) {
    const float p = __expf(vals[v] - m) * inv * 0.9f + 0.1f / V_;  // (1-UW)*soft + UW/V
    vals[v] = p;
    if (p > bestv || (p == bestv && v < besti)) { bestv = p; besti = v; }
  }
  red[t] = bestv; redi[t] = besti; __syncthreads();
  for (int o = 128; o > 0; o >>= 1) {
    if (t < o) {
      if (red[t + o] > red[t] || (red[t + o] == red[t] && redi[t + o] < redi[t])) {
        red[t] = red[t + o]; redi[t] = redi[t + o];
      }
    }
    __syncthreads();
  }
  const int am = redi[0]; __syncthreads();

  for (int v = t; v < V_; v += 256) {
    const float p = vals[v];
    const float hard = (v == am) ? 1.f : 0.f;
    const float st = (hard - p) + p;   // straight-through value (fp-faithful)
    st_f[(size_t)row * V_ + v] = st;
    st_bf[(size_t)row * V_ + v] = (bf16_t)st;
  }
}

// ---------------------------------------------------------------------------
// Pack output: [sos onehot | st (B,C,V) | eos onehot] -> (B, 16, V)
// ---------------------------------------------------------------------------
__global__ __launch_bounds__(256) void write_out_k(const float* __restrict__ st_f,
                                                   float* __restrict__ out) {
  const int idx = blockIdx.x * 256 + threadIdx.x;  // B*16*V exactly
  const int v = idx % V_, rp = idx / V_, p = rp % (C_ + 2), b = rp / (C_ + 2);
  float val;
  if (p == 0)            val = (v == 0) ? 1.f : 0.f;   // SOS
  else if (p == C_ + 1)  val = (v == 1) ? 1.f : 0.f;   // EOS
  else                   val = st_f[(size_t)(b * C_ + (p - 1)) * V_ + v];
  out[idx] = val;
}

// ---------------------------------------------------------------------------
// Host orchestration
// ---------------------------------------------------------------------------
namespace {
struct WsAlloc {
  char* p;
  template <class T> T* take(size_t n) {
    T* r = (T*)p;
    p += ((n * sizeof(T)) + 255) & ~(size_t)255;
    return r;
  }
};

inline void launch_gemm(const bf16_t* A, const bf16_t* Bp, float* C, const float* bias,
                        int M, int N, int K, bool add_res, hipStream_t s) {
  dim3 g(N / BN, (M + BM - 1) / BM), blk(256);
  if (add_res) gemm_bf16_wmma_k<true><<<g, blk, 0, s>>>(A, Bp, C, bias, M, N, K);
  else         gemm_bf16_wmma_k<false><<<g, blk, 0, s>>>(A, Bp, C, bias, M, N, K);
}

inline void conv_bf16(const float* src, bf16_t* dst, size_t n, hipStream_t s) {
  f32_to_bf16_k<<<(unsigned)((n + 255) / 256), 256, 0, s>>>(src, dst, (int)n);
}

inline void pack_b(const float* src, bf16_t* dst, int K, int N, hipStream_t s) {
  pack_b_bf16_k<<<(unsigned)(((size_t)K * N + 255) / 256), 256, 0, s>>>(src, dst, K, N);
}
}  // namespace

extern "C" void kernel_launch(void* const* d_in, const int* in_sizes, int n_in,
                              void* d_out, int out_size, void* d_ws, size_t ws_size,
                              hipStream_t stream) {
  (void)in_sizes; (void)n_in; (void)out_size; (void)ws_size;

  const float* prototypes = (const float*)d_in[0];
  const float* query      = (const float*)d_in[1];
  const float* ca_wq      = (const float*)d_in[2];
  const float* ca_wk      = (const float*)d_in[3];
  const float* ca_wv      = (const float*)d_in[4];
  const float* ca_wo      = (const float*)d_in[5];
  const float* ln1_w      = (const float*)d_in[6];
  const float* ln1_b      = (const float*)d_in[7];
  const float* qkv_w      = (const float*)d_in[8];
  const float* attn_out_w = (const float*)d_in[9];
  const float* rel_bias   = (const float*)d_in[10];
  const float* ln2_w      = (const float*)d_in[11];
  const float* ln2_b      = (const float*)d_in[12];
  const float* mlp_w1     = (const float*)d_in[13];
  const float* mlp_w2     = (const float*)d_in[14];
  const float* lnf_w      = (const float*)d_in[15];
  const float* lnf_b      = (const float*)d_in[16];
  const float* vocab_w    = (const float*)d_in[17];
  const float* vocab_b    = (const float*)d_in[18];
  const float* pos_emb    = (const float*)d_in[19];
  const float* tok_emb    = (const float*)d_in[20];

  WsAlloc ws{(char*)d_ws};

  // --- bf16 weight mirrors, packed into WMMA B-fragment layout ---
  bf16_t* wq_bf    = ws.take<bf16_t>((size_t)D_ * D_);
  bf16_t* wk_bf    = ws.take<bf16_t>((size_t)D_ * D_);
  bf16_t* wv_bf    = ws.take<bf16_t>((size_t)D_ * D_);
  bf16_t* wo_bf    = ws.take<bf16_t>((size_t)D_ * D_);
  bf16_t* query_bf = ws.take<bf16_t>((size_t)BOS_ * D_);    // A-side: plain
  bf16_t* proto_bf = ws.take<bf16_t>((size_t)B_ * P_ * D_); // A-side: plain
  bf16_t* qkvw_bf  = ws.take<bf16_t>((size_t)L_ * D_ * 3 * D_);
  bf16_t* aow_bf   = ws.take<bf16_t>((size_t)L_ * D_ * D_);
  bf16_t* m1_bf    = ws.take<bf16_t>((size_t)L_ * D_ * 2 * HID_);
  bf16_t* m2_bf    = ws.take<bf16_t>((size_t)L_ * HID_ * D_);
  bf16_t* voc_bf   = ws.take<bf16_t>((size_t)D_ * V_);
  bf16_t* tok_bf   = ws.take<bf16_t>((size_t)V_ * D_);

  // --- activations ---
  float*  bias_f   = ws.take<float>((size_t)L_ * H_ * S_ * S_);
  float*  qh_f     = ws.take<float>((size_t)BOS_ * D_);
  float*  kh_f     = ws.take<float>((size_t)B_ * P_ * D_);
  float*  vh_f     = ws.take<float>((size_t)B_ * P_ * D_);
  bf16_t* caout_bf = ws.take<bf16_t>((size_t)B_ * BOS_ * D_);
  float*  bos_f    = ws.take<float>((size_t)B_ * BOS_ * D_);
  float*  x_f      = ws.take<float>((size_t)B_ * S_ * D_);
  bf16_t* h_bf     = ws.take<bf16_t>((size_t)B_ * S_ * D_);
  float*  qkv_f    = ws.take<float>((size_t)B_ * S_ * 3 * D_);
  bf16_t* ao_bf    = ws.take<bf16_t>((size_t)B_ * S_ * D_);
  float*  u_f      = ws.take<float>((size_t)B_ * S_ * 2 * HID_);
  bf16_t* g_bf     = ws.take<bf16_t>((size_t)B_ * S_ * HID_);
  bf16_t* xf_bf    = ws.take<bf16_t>((size_t)B_ * C_ * D_);
  float*  log_f    = ws.take<float>((size_t)B_ * C_ * V_);
  float*  st_f     = ws.take<float>((size_t)B_ * C_ * V_);
  bf16_t* st_bf    = ws.take<bf16_t>((size_t)B_ * C_ * V_);
  float*  spoken_f = ws.take<float>((size_t)B_ * C_ * D_);

  // --- one-time (per launch) weight conversion + fragment packing ---
  pack_b(ca_wq, wq_bf, D_, D_, stream);
  pack_b(ca_wk, wk_bf, D_, D_, stream);
  pack_b(ca_wv, wv_bf, D_, D_, stream);
  pack_b(ca_wo, wo_bf, D_, D_, stream);
  conv_bf16(query, query_bf, (size_t)BOS_ * D_, stream);
  conv_bf16(prototypes, proto_bf, (size_t)B_ * P_ * D_, stream);
  for (int l = 0; l < L_; ++l) {
    pack_b(qkv_w + (size_t)l * D_ * 3 * D_, qkvw_bf + (size_t)l * D_ * 3 * D_, D_, 3 * D_, stream);
    pack_b(attn_out_w + (size_t)l * D_ * D_, aow_bf + (size_t)l * D_ * D_, D_, D_, stream);
    pack_b(mlp_w1 + (size_t)l * D_ * 2 * HID_, m1_bf + (size_t)l * D_ * 2 * HID_, D_, 2 * HID_, stream);
    pack_b(mlp_w2 + (size_t)l * HID_ * D_, m2_bf + (size_t)l * HID_ * D_, HID_, D_, stream);
  }
  pack_b(vocab_w, voc_bf, D_, V_, stream);
  pack_b(tok_emb, tok_bf, V_, D_, stream);

  bias_pre_k<<<(L_ * H_ * S_ * S_ + 255) / 256, 256, 0, stream>>>(rel_bias, bias_f);

  // --- cross-attention -> bos (B, BOS, D) ---
  launch_gemm(query_bf, wq_bf, qh_f, nullptr, BOS_, D_, D_, false, stream);
  launch_gemm(proto_bf, wk_bf, kh_f, nullptr, B_ * P_, D_, D_, false, stream);
  launch_gemm(proto_bf, wv_bf, vh_f, nullptr, B_ * P_, D_, D_, false, stream);
  ca_attn_k<<<B_ * H_, 256, 0, stream>>>(qh_f, kh_f, vh_f, caout_bf);
  launch_gemm(caout_bf, wo_bf, bos_f, nullptr, B_ * BOS_, D_, D_, false, stream);

  const int ROWS = B_ * S_;  // 1920

  // --- 14-step autoregressive scan ---
  for (int i = 0; i < C_; ++i) {
    build_x_k<<<(B_ * S_ * D_) / 256, 256, 0, stream>>>(x_f, bos_f, spoken_f, pos_emb, i);

    for (int l = 0; l < L_; ++l) {
      ln_bf16_k<<<ROWS, 256, 0, stream>>>(x_f, ln1_w + (size_t)l * D_, ln1_b + (size_t)l * D_,
                                          h_bf, S_, S_, 0);
      launch_gemm(h_bf, qkvw_bf + (size_t)l * D_ * 3 * D_, qkv_f, nullptr,
                  ROWS, 3 * D_, D_, false, stream);
      attn_k<<<B_ * H_, 256, 0, stream>>>(qkv_f, bias_f + (size_t)l * H_ * S_ * S_, ao_bf);
      launch_gemm(ao_bf, aow_bf + (size_t)l * D_ * D_, x_f, nullptr,
                  ROWS, D_, D_, true, stream);                       // residual add
      ln_bf16_k<<<ROWS, 256, 0, stream>>>(x_f, ln2_w + (size_t)l * D_, ln2_b + (size_t)l * D_,
                                          h_bf, S_, S_, 0);
      launch_gemm(h_bf, m1_bf + (size_t)l * D_ * 2 * HID_, u_f, nullptr,
                  ROWS, 2 * HID_, D_, false, stream);
      swiglu_k<<<(ROWS * HID_) / 256, 256, 0, stream>>>(u_f, g_bf, ROWS);
      launch_gemm(g_bf, m2_bf + (size_t)l * HID_ * D_, x_f, nullptr,
                  ROWS, D_, HID_, true, stream);                     // residual add
    }

    // final LN over content rows only -> (B*C, D) bf16
    ln_bf16_k<<<B_ * C_, 256, 0, stream>>>(x_f, lnf_w, lnf_b, xf_bf, S_, C_, BOS_);
    launch_gemm(xf_bf, voc_bf, log_f, vocab_b, B_ * C_, V_, D_, false, stream);
    sample_k<<<B_ * C_, 256, 0, stream>>>(log_f, st_f, st_bf, i);
    launch_gemm(st_bf, tok_bf, spoken_f, nullptr, B_ * C_, D_, V_, false, stream);
  }

  write_out_k<<<(B_ * (C_ + 2) * V_) / 256, 256, 0, stream>>>(st_f, (float*)d_out);
}
